// msstgcn_91259465105917
// MI455X (gfx1250) — matrix operand order repaired and checked by reading, not compiled, not run on hardware
//
#include <hip/hip_runtime.h>
#include <hip/hip_bf16.h>

typedef _Float16 half_t;
typedef __attribute__((ext_vector_type(16))) _Float16 v16h;
typedef __attribute__((ext_vector_type(8)))  _Float16 v8h;
typedef __attribute__((ext_vector_type(8)))  float    v8f;

#define SHUF16(lo, hi) __builtin_shufflevector((lo), (hi), 0,1,2,3,4,5,6,7,8,9,10,11,12,13,14,15)

#define EPSF 1e-5f
#define SDIM 204800   // N*T*V = 8*1024*25
#define NT   8192     // N*T

// ---------------------------------------------------------------------------
// Kernel 0: precompute f16 weights + fused BN affine coefficients
// ---------------------------------------------------------------------------
__global__ void prep_coefs(const float* __restrict__ gw, const float* __restrict__ tw,
                           const float* __restrict__ tb,
                           const float* __restrict__ b1g, const float* __restrict__ b1b,
                           const float* __restrict__ b1m, const float* __restrict__ b1v,
                           const float* __restrict__ b2g, const float* __restrict__ b2b,
                           const float* __restrict__ b2m, const float* __restrict__ b2v,
                           const float* __restrict__ b3g, const float* __restrict__ b3b,
                           const float* __restrict__ b3m, const float* __restrict__ b3v,
                           const float* __restrict__ b4g, const float* __restrict__ b4b,
                           const float* __restrict__ b4m, const float* __restrict__ b4v,
                           const float* __restrict__ dg, const float* __restrict__ db,
                           const float* __restrict__ dm, const float* __restrict__ dv,
                           half_t* __restrict__ GW, half_t* __restrict__ TW,
                           float* __restrict__ A12, float* __restrict__ C12,
                           float* __restrict__ A3,  float* __restrict__ C3,
                           float* __restrict__ A4,  float* __restrict__ C4,
                           float* __restrict__ AD,  float* __restrict__ CD) {
  int i = blockIdx.x * blockDim.x + threadIdx.x;
  // tcn weights: (L,F,C,5) f32 -> [L][K=5][F][C] f16
  if (i < 10 * 5 * 64 * 64) {
    int l = i / 20480, r = i % 20480;
    int k = r / 4096;  r = r % 4096;
    int f = r / 64, c = r % 64;
    TW[i] = (half_t)tw[(((size_t)l * 64 + f) * 64 + c) * 5 + k];
  }
  // gcn weights: (L,64,64) f32 -> f16 (same layout)
  if (i < 10 * 64 * 64) GW[i] = (half_t)gw[i];
  // fused BN affines, per (layer, channel)
  if (i < 640) {
    float k1 = b1g[i] * rsqrtf(b1v[i] + EPSF);
    float k2 = b2g[i] * rsqrtf(b2v[i] + EPSF);
    float k3 = b3g[i] * rsqrtf(b3v[i] + EPSF);
    float k4 = b4g[i] * rsqrtf(b4v[i] + EPSF);
    A12[i] = k1 * k2;
    C12[i] = (b1b[i] - b1m[i] * k1 - b2m[i]) * k2 + b2b[i];
    A3[i]  = k3;
    C3[i]  = (tb[i] - b3m[i]) * k3 + b3b[i];
    A4[i]  = k4;
    C4[i]  = b4b[i] - b4m[i] * k4;
  }
  // data_bn affine over channel j = v*2+c (50 channels)
  if (i < 50) {
    float kd = dg[i] * rsqrtf(dv[i] + EPSF);
    AD[i] = kd;
    CD[i] = db[i] - dm[i] * kd;
  }
}

// ---------------------------------------------------------------------------
// Kernel 1: data_bn + view-scramble + conv1 (C=2 -> F=64), write Y f16 [S][64]
// ---------------------------------------------------------------------------
__global__ void prep_input(const float* __restrict__ x, const float* __restrict__ w1,
                           const float* __restrict__ bias1,
                           const float* __restrict__ AD, const float* __restrict__ CD,
                           half_t* __restrict__ Y) {
  int s = blockIdx.x * blockDim.x + threadIdx.x;   // s = n*T*V + t*V + v, grid exact
  int n = s / 25600, r = s % 25600;
  int t = r / 25, v = r % 25;
  float a[2];
#pragma unroll
  for (int c = 0; c < 2; ++c) {
    int i  = c * 25600 + t * 25 + v;   // flat (V,T,C) index of torch .view scramble
    int vv = i >> 11;                  // / (T*C) = 2048
    int r2 = i & 2047;
    int tt = r2 >> 1;
    int cc = r2 & 1;
    int j  = vv * 2 + cc;              // data_bn channel
    float xv = x[(((size_t)n * 2 + cc) * 1024 + tt) * 25 + vv];
    a[c] = xv * AD[j] + CD[j];
  }
  half_t* yr = Y + (size_t)s * 64;
#pragma unroll
  for (int f0 = 0; f0 < 64; f0 += 8) {
    v8h o;
#pragma unroll
    for (int j = 0; j < 8; ++j) {
      int f = f0 + j;
      o[j] = (half_t)(w1[f * 2] * a[0] + w1[f * 2 + 1] * a[1] + bias1[f]);
    }
    *(v8h*)(yr + f0) = o;
  }
}

// ---------------------------------------------------------------------------
// Kernel 2: gcn 64x64 channel GEMM via WMMA.  Y(f16) -> Z(f16), + gcn bias.
// Tile: M=out-ch (4x16), N=16 spatial rows per wave, K=64 (2 steps of 32).
// ---------------------------------------------------------------------------
__global__ __launch_bounds__(256) void gcn_gemm(const half_t* __restrict__ Y,
                                                const half_t* __restrict__ GW,
                                                const float* __restrict__ gb,
                                                half_t* __restrict__ Z, int layer) {
  __shared__ __align__(16) half_t wlds[64 * 64];
  int tid = threadIdx.x;
  { // stage layer weights into LDS (8 KB)
    const uint4* src = (const uint4*)(GW + (size_t)layer * 4096);
    uint4* dst = (uint4*)wlds;
    for (int i = tid; i < 512; i += 256) dst[i] = src[i];
  }
  __syncthreads();

  int wave = tid >> 5, lane = tid & 31;
  int ln = lane & 15, lh = lane >> 4;
  int s = blockIdx.x * 128 + wave * 16 + ln;          // spatial row for this lane's column
  const half_t* yrow = Y + (size_t)s * 64;

  v8f acc[4] = {};
#pragma unroll
  for (int ks = 0; ks < 2; ++ks) {
    int kb = ks * 32 + lh * 16;
    v8h blo = *(const v8h*)(yrow + kb);
    v8h bhi = *(const v8h*)(yrow + kb + 8);
    v16h bfrag = SHUF16(blo, bhi);
#pragma unroll
    for (int mt = 0; mt < 4; ++mt) {
      const half_t* wr = wlds + (mt * 16 + ln) * 64 + ks * 32 + lh * 16;
      v8h alo = *(const v8h*)(wr);
      v8h ahi = *(const v8h*)(wr + 8);
      v16h afrag = SHUF16(alo, ahi);
      acc[mt] = __builtin_amdgcn_wmma_f32_16x16x32_f16(
          false, afrag, false, bfrag, (short)0, acc[mt], false, false);
    }
  }
  // epilogue: + gcn bias, pack 8 consecutive channels -> one b128 store
#pragma unroll
  for (int mt = 0; mt < 4; ++mt) {
    int ch0 = mt * 16 + lh * 8;
    const float* gbp = gb + layer * 64 + ch0;
    v8h o;
#pragma unroll
    for (int j = 0; j < 8; ++j) o[j] = (half_t)(acc[mt][j] + gbp[j]);
    *(v8h*)(Z + (size_t)s * 64 + ch0) = o;
  }
}

// ---------------------------------------------------------------------------
// Kernel 3: graph contraction over V (25x25) + fused bn1∘bn2 + ReLU, in-place on Z
// One block per (n,t); the 25x64 tile is self-contained.
// ---------------------------------------------------------------------------
__global__ __launch_bounds__(256) void a_contract(half_t* __restrict__ Z,
                                                  const float* __restrict__ Ag,
                                                  const float* __restrict__ A12,
                                                  const float* __restrict__ C12, int layer) {
  __shared__ __align__(16) half_t zt[25 * 64];
  __shared__ float Af[625];
  int tid = threadIdx.x;
  size_t sb = (size_t)blockIdx.x * 25;
  { // stage 25x64 f16 tile (200 x uint4) and A (625 f32)
    const uint4* src = (const uint4*)(Z + sb * 64);
    uint4* dst = (uint4*)zt;
    for (int i = tid; i < 200; i += 256) dst[i] = src[i];
    for (int i = tid; i < 625; i += 256) Af[i] = Ag[i];
  }
  __syncthreads();
  const float* a12 = A12 + layer * 64;
  const float* c12 = C12 + layer * 64;
  for (int idx = tid; idx < 1600; idx += 256) {
    int w = idx >> 6, c = idx & 63;
    float sum = 0.f;
#pragma unroll
    for (int v = 0; v < 25; ++v) sum += (float)zt[v * 64 + c] * Af[v * 25 + w];
    float val = sum * a12[c] + c12[c];
    val = fmaxf(val, 0.f);
    Z[sb * 64 + idx] = (half_t)val;
  }
}

// ---------------------------------------------------------------------------
// Kernel 4: tcn = 5 shifted 64x64 GEMMs (dilated, zero-pad via data predication)
// + fused bn3/ReLU/bn4 + residual, in-place update of Y.
// ---------------------------------------------------------------------------
__global__ __launch_bounds__(256) void tcn_gemm(const half_t* __restrict__ Zr,
                                                half_t* __restrict__ Y,
                                                const half_t* __restrict__ TW,
                                                const float* __restrict__ A3,
                                                const float* __restrict__ C3,
                                                const float* __restrict__ A4,
                                                const float* __restrict__ C4,
                                                int layer, int dil) {
  __shared__ __align__(16) half_t wlds[5 * 64 * 64];   // 40 KB
  int tid = threadIdx.x;
  { // stage the layer's 5 tap weight matrices
    const uint4* src = (const uint4*)(TW + (size_t)layer * 20480);
    uint4* dst = (uint4*)wlds;
    for (int i = tid; i < 2560; i += 256) dst[i] = src[i];
  }
  __syncthreads();

  int wave = tid >> 5, lane = tid & 31;
  int ln = lane & 15, lh = lane >> 4;
  int s = blockIdx.x * 128 + wave * 16 + ln;
  int t = (s / 25) % 1024;
  __builtin_prefetch(Y + (size_t)s * 64, 0, 1);        // residual row -> global_prefetch

  v8f acc[4] = {};
  v8h z8 = {};
#pragma unroll
  for (int k = 0; k < 5; ++k) {
    int tsh = (k - 2) * dil;
    bool valid = (unsigned)(t + tsh) < 1024u;
    const half_t* zrow = Zr + ((long)s + (long)tsh * 25) * 64;
#pragma unroll
    for (int ks = 0; ks < 2; ++ks) {
      int kb = ks * 32 + lh * 16;
      v8h blo = z8, bhi = z8;
      if (valid) {
        blo = *(const v8h*)(zrow + kb);
        bhi = *(const v8h*)(zrow + kb + 8);
      }
      v16h bfrag = SHUF16(blo, bhi);
#pragma unroll
      for (int mt = 0; mt < 4; ++mt) {
        const half_t* wr = wlds + k * 4096 + (mt * 16 + ln) * 64 + ks * 32 + lh * 16;
        v8h alo = *(const v8h*)(wr);
        v8h ahi = *(const v8h*)(wr + 8);
        v16h afrag = SHUF16(alo, ahi);
        acc[mt] = __builtin_amdgcn_wmma_f32_16x16x32_f16(
            false, afrag, false, bfrag, (short)0, acc[mt], false, false);
      }
    }
  }
  // epilogue: (h+bias)->bn3 -> ReLU -> bn4 -> + residual, write back f16
#pragma unroll
  for (int mt = 0; mt < 4; ++mt) {
    int ch0 = mt * 16 + lh * 8;
    const float* a3 = A3 + layer * 64 + ch0;
    const float* c3 = C3 + layer * 64 + ch0;
    const float* a4 = A4 + layer * 64 + ch0;
    const float* c4 = C4 + layer * 64 + ch0;
    v8h res = *(const v8h*)(Y + (size_t)s * 64 + ch0);
    v8h o;
#pragma unroll
    for (int j = 0; j < 8; ++j) {
      float h = acc[mt][j];
      h = h * a3[j] + c3[j];
      h = fmaxf(h, 0.f);
      h = h * a4[j] + c4[j];
      h += (float)res[j];
      o[j] = (half_t)h;
    }
    *(v8h*)(Y + (size_t)s * 64 + ch0) = o;
  }
}

// ---------------------------------------------------------------------------
// Kernel 5: mean over V, 2x64 output conv, mask, 2-class softmax. Wave/(n,t).
// ---------------------------------------------------------------------------
__global__ __launch_bounds__(256) void head_k(const half_t* __restrict__ Y,
                                              const float* __restrict__ ow,
                                              const float* __restrict__ ob,
                                              const float* __restrict__ maskp,
                                              float* __restrict__ out) {
  int tid = threadIdx.x, wave = tid >> 5, lane = tid & 31;
  int nt = blockIdx.x * 8 + wave;                  // < 8192
  int n = nt >> 10, t = nt & 1023;
  const half_t* yb = Y + (size_t)nt * 25 * 64;
  float s0 = 0.f, s1 = 0.f;
  for (int v = 0; v < 25; ++v) {
    s0 += (float)yb[v * 64 + lane];
    s1 += (float)yb[v * 64 + 32 + lane];
  }
  s0 *= (1.f / 25.f);
  s1 *= (1.f / 25.f);
  float l0 = ow[lane] * s0 + ow[32 + lane] * s1;
  float l1 = ow[64 + lane] * s0 + ow[96 + lane] * s1;
#pragma unroll
  for (int off = 16; off > 0; off >>= 1) {
    l0 += __shfl_xor(l0, off, 32);
    l1 += __shfl_xor(l1, off, 32);
  }
  if (lane == 0) {
    float m = maskp[(size_t)n * 2048 + t];         // mask[n,0,t]
    l0 = (l0 + ob[0]) * m;
    l1 = (l1 + ob[1]) * m;
    float mx = fmaxf(l0, l1);
    float e0 = expf(l0 - mx), e1 = expf(l1 - mx);
    float inv = 1.f / (e0 + e1);
    out[((size_t)n * 2 + 0) * 1024 + t] = e0 * inv;
    out[((size_t)n * 2 + 1) * 1024 + t] = e1 * inv;
  }
}

// ---------------------------------------------------------------------------
extern "C" void kernel_launch(void* const* d_in, const int* in_sizes, int n_in,
                              void* d_out, int out_size, void* d_ws, size_t ws_size,
                              hipStream_t stream) {
  (void)in_sizes; (void)n_in; (void)out_size; (void)ws_size;
  const float* x    = (const float*)d_in[0];
  const float* mask = (const float*)d_in[1];
  const float* Ag   = (const float*)d_in[2];
  const float* dg   = (const float*)d_in[3];
  const float* db   = (const float*)d_in[4];
  const float* dm   = (const float*)d_in[5];
  const float* dv   = (const float*)d_in[6];
  const float* w1   = (const float*)d_in[7];
  const float* b1   = (const float*)d_in[8];
  const float* gw   = (const float*)d_in[9];
  const float* gb   = (const float*)d_in[10];
  const float* b1g  = (const float*)d_in[11];
  const float* b1b  = (const float*)d_in[12];
  const float* b1m  = (const float*)d_in[13];
  const float* b1v  = (const float*)d_in[14];
  const float* tw   = (const float*)d_in[15];
  const float* tb   = (const float*)d_in[16];
  const float* b2g  = (const float*)d_in[17];
  const float* b2b  = (const float*)d_in[18];
  const float* b2m  = (const float*)d_in[19];
  const float* b2v  = (const float*)d_in[20];
  const float* b3g  = (const float*)d_in[21];
  const float* b3b  = (const float*)d_in[22];
  const float* b3m  = (const float*)d_in[23];
  const float* b3v  = (const float*)d_in[24];
  const float* b4g  = (const float*)d_in[25];
  const float* b4b  = (const float*)d_in[26];
  const float* b4m  = (const float*)d_in[27];
  const float* b4v  = (const float*)d_in[28];
  const float* ow   = (const float*)d_in[29];
  const float* ob   = (const float*)d_in[30];

  char* ws = (char*)d_ws;
  half_t* Y   = (half_t*)(ws);                       // 26,214,400 B
  half_t* Z   = (half_t*)(ws + 26214400);            // 26,214,400 B
  half_t* GW  = (half_t*)(ws + 52428800);            //     81,920 B
  half_t* TW  = (half_t*)(ws + 52510720);            //    409,600 B
  float*  A12 = (float*) (ws + 52920320);
  float*  C12 = (float*) (ws + 52922880);
  float*  A3  = (float*) (ws + 52925440);
  float*  C3  = (float*) (ws + 52928000);
  float*  A4  = (float*) (ws + 52930560);
  float*  C4  = (float*) (ws + 52933120);
  float*  AD  = (float*) (ws + 52935680);
  float*  CD  = (float*) (ws + 52935936);

  prep_coefs<<<800, 256, 0, stream>>>(gw, tw, tb,
      b1g, b1b, b1m, b1v, b2g, b2b, b2m, b2v,
      b3g, b3b, b3m, b3v, b4g, b4b, b4m, b4v,
      dg, db, dm, dv,
      GW, TW, A12, C12, A3, C3, A4, C4, AD, CD);

  prep_input<<<800, 256, 0, stream>>>(x, w1, b1, AD, CD, Y);

  static const int DIL[10] = {1, 2, 4, 8, 16, 32, 64, 128, 256, 512};
  for (int l = 0; l < 10; ++l) {
    gcn_gemm<<<1600, 256, 0, stream>>>(Y, GW, gb, Z, l);
    a_contract<<<NT, 256, 0, stream>>>(Z, Ag, A12, C12, l);
    tcn_gemm<<<1600, 256, 0, stream>>>(Z, Y, TW, A3, C3, A4, C4, l, DIL[l]);
  }

  head_k<<<1024, 256, 0, stream>>>(Y, ow, ob, mask, (float*)d_out);
}
